// BANLayer_47253230190789
// MI455X (gfx1250) — matrix operand order repaired and checked by reading, not compiled, not run on hardware
//
#include <hip/hip_runtime.h>
#include <hip/hip_bf16.h>

#define B_     64
#define VN_    256
#define QN_    1024
#define V_DIM_ 128
#define Q_DIM_ 1280
#define H_DIM_ 256
#define H_OUT_ 2
#define KGRP_  3
#define HK_    768
#define LN_EPS 1e-5f

typedef __attribute__((ext_vector_type(16))) __bf16 v16bf;
typedef __attribute__((ext_vector_type(8)))  __bf16 v8bf;
typedef __attribute__((ext_vector_type(8)))  float  v8f;

#define LDT 40  // LDS row stride in bf16 elems: 32 + 8 pad (no bank conflicts, keeps 16B align)

#define AS1 __attribute__((address_space(1)))
#define AS3 __attribute__((address_space(3)))

// ---- CDNA5 async load-to-LDS (ASYNCcnt path), guarded so either toolchain compiles ----
#if defined(__has_builtin)
#if __has_builtin(__builtin_amdgcn_global_load_async_to_lds_b128)
#define HAVE_ASYNC_LDS 1
#endif
#endif

#if defined(__has_builtin)
#if __has_builtin(__builtin_amdgcn_s_wait_asynccnt)
#define WAIT_ASYNC() __builtin_amdgcn_s_wait_asynccnt(0)
#else
#define WAIT_ASYNC() asm volatile("s_wait_asynccnt 0x0" ::: "memory")
#endif
#else
#define WAIT_ASYNC() asm volatile("s_wait_asynccnt 0x0" ::: "memory")
#endif

#if defined(HAVE_ASYNC_LDS)
// Param types per hipcc diagnostic: (int-vec16B __device__* src, int-vec16B __shared__* dst, Ii, Ii)
typedef int v4i_ __attribute__((vector_size(16)));
// generic->as(1): bit-identical 64-bit value; generic->as(3): LDS offset lives in addr[31:0]
__device__ __forceinline__ void async_copy_b128(const void* gsrc, void* ldst) {
    __builtin_amdgcn_global_load_async_to_lds_b128(
        (AS1 v4i_*)(uintptr_t)gsrc,
        (AS3 v4i_*)(unsigned)(uintptr_t)ldst, 0, 0);
}
#endif

// ---------------- WMMA fragment loaders (ISA 7.12.2 layouts) ----------------
// A 16x32 bf16: lane L<16 -> row M=L, K = {0..7, 16..23}; lane L>=16 -> row M=L-16, K = {8..15, 24..31}
__device__ __forceinline__ v16bf frag_a(const __bf16* base, int lane) {
    int r  = lane & 15;
    int kb = (lane >> 4) << 3;  // 0 or 8
    const __bf16* p = base + r * LDT + kb;
    v8bf lo = *(const v8bf*)(p);        // K = kb .. kb+7
    v8bf hi = *(const v8bf*)(p + 16);   // K = kb+16 .. kb+23
    v16bf f;
#pragma unroll
    for (int i = 0; i < 8; ++i) { f[i] = lo[i]; f[i + 8] = hi[i]; }
    return f;
}

// B 32x16 bf16 staged as [N][K] row-major: lane L -> col N=L&15, K = (L<16 ? {0..7, 16..23} : {8..15, 24..31})
__device__ __forceinline__ v16bf frag_b(const __bf16* base, int lane) {
    int n  = lane & 15;
    int kb = (lane >> 4) << 3;  // 0 or 8
    const __bf16* p = base + n * LDT + kb;
    v8bf lo = *(const v8bf*)(p);
    v8bf hi = *(const v8bf*)(p + 16);
    v16bf f;
#pragma unroll
    for (int i = 0; i < 8; ++i) { f[i] = lo[i]; f[i + 8] = hi[i]; }
    return f;
}

__device__ __forceinline__ float block_reduce_sum(float v, float* red, int tid) {
    red[tid] = v; __syncthreads();
    for (int s = 128; s > 0; s >>= 1) {
        if (tid < s) red[tid] += red[tid + s];
        __syncthreads();
    }
    float r = red[0]; __syncthreads();
    return r;
}

__device__ __forceinline__ float block_reduce_max(float v, float* red, int tid) {
    red[tid] = v; __syncthreads();
    for (int s = 128; s > 0; s >>= 1) {
        if (tid < s) red[tid] = fmaxf(red[tid], red[tid + s]);
        __syncthreads();
    }
    float r = red[0]; __syncthreads();
    return r;
}

// ---------------- K0: sum of squares of weight matrices (weight_norm) ----------------
__global__ __launch_bounds__(256) void ssq_kernel(const float* __restrict__ v_w,
                                                  const float* __restrict__ q_w,
                                                  float* __restrict__ ssq) {
    __shared__ float red[256];
    const float* w = (blockIdx.x == 0) ? v_w : q_w;
    int n = (blockIdx.x == 0) ? (HK_ * V_DIM_) : (HK_ * Q_DIM_);
    float s = 0.f;
    for (int i = threadIdx.x; i < n; i += 256) { float t = w[i]; s += t * t; }
    float tot = block_reduce_sum(s, red, threadIdx.x);
    if (threadIdx.x == 0) ssq[blockIdx.x] = tot;
}

// ---------------- K1/K2: projection  out[m,n] = relu(scale * X[m,:].W[n,:] + b[n]) * mask[m] --------
// X: [M x Kdim] fp32 row-major, W: [768 x Kdim] fp32 row-major (already [N][K]).  out: bf16 [M x 768].
// Double-buffered LDS ping-pong: global loads of tile kt+1 overlap WMMAs on tile kt.
__global__ __launch_bounds__(256) void proj_kernel(const float* __restrict__ X,
                                                   const float* __restrict__ W,
                                                   const float* __restrict__ bias,
                                                   const int*   __restrict__ mask,
                                                   const float* __restrict__ g,
                                                   const float* __restrict__ ssq,
                                                   int Kdim,
                                                   __bf16* __restrict__ out) {
    __shared__ __bf16 ldsA[2][128 * LDT];
    __shared__ __bf16 ldsB[2][128 * LDT];
    const int m0 = blockIdx.x * 128;
    const int n0 = blockIdx.y * 128;
    const int tid = threadIdx.x;
    const int lane = tid & 31, wid = tid >> 5;
    const int wm = wid >> 2, wn = wid & 3;           // 2 x 4 wave grid
    const int r = tid >> 1;                          // 0..127: tile row handled by this thread
    const int half = (tid & 1) * 16;                 // 0 or 16: K sub-chunk

    v8f acc[4][2];
#pragma unroll
    for (int i = 0; i < 4; ++i)
#pragma unroll
        for (int j = 0; j < 2; ++j) acc[i][j] = (v8f){0.f,0.f,0.f,0.f,0.f,0.f,0.f,0.f};

    float fa[16], fb[16];
    const int nk = Kdim >> 5;

    auto issue = [&](int kt) {
        const float* pa = X + (size_t)(m0 + r) * Kdim + kt * 32 + half;
        const float* pb = W + (size_t)(n0 + r) * Kdim + kt * 32 + half;
        if (kt + 1 < nk) { __builtin_prefetch(pa + 32, 0, 1); __builtin_prefetch(pb + 32, 0, 1); }
#pragma unroll
        for (int i = 0; i < 16; ++i) { fa[i] = pa[i]; fb[i] = pb[i]; }
    };
    auto commit = [&](int buf) {
        v8bf a0, a1, b0, b1;
#pragma unroll
        for (int i = 0; i < 8; ++i) {
            a0[i] = (__bf16)fa[i];  a1[i] = (__bf16)fa[8 + i];
            b0[i] = (__bf16)fb[i];  b1[i] = (__bf16)fb[8 + i];
        }
        *(v8bf*)&ldsA[buf][r * LDT + half]     = a0;  *(v8bf*)&ldsA[buf][r * LDT + half + 8] = a1;
        *(v8bf*)&ldsB[buf][r * LDT + half]     = b0;  *(v8bf*)&ldsB[buf][r * LDT + half + 8] = b1;
    };

    issue(0); commit(0);
    __syncthreads();
    for (int kt = 0; kt < nk; ++kt) {
        const int buf = kt & 1;
        const bool more = (kt + 1) < nk;
        if (more) issue(kt + 1);
        v16bf af[4], bfr[2];
#pragma unroll
        for (int ti = 0; ti < 4; ++ti) af[ti] = frag_a(&ldsA[buf][(wm * 64 + ti * 16) * LDT], lane);
#pragma unroll
        for (int tj = 0; tj < 2; ++tj) bfr[tj] = frag_b(&ldsB[buf][(wn * 32 + tj * 16) * LDT], lane);
#pragma unroll
        for (int ti = 0; ti < 4; ++ti)
#pragma unroll
            for (int tj = 0; tj < 2; ++tj)
                acc[ti][tj] = __builtin_amdgcn_wmma_f32_16x16x32_bf16(
                    false, af[ti], false, bfr[tj], (short)0, acc[ti][tj], false, false);
        if (more) commit(buf ^ 1);
        __syncthreads();
    }

    const float scale = g[0] * rsqrtf(ssq[0]);
#pragma unroll
    for (int ti = 0; ti < 4; ++ti)
#pragma unroll
        for (int tj = 0; tj < 2; ++tj)
#pragma unroll
            for (int j = 0; j < 8; ++j) {
                int m = m0 + wm * 64 + ti * 16 + ((lane >> 4) << 3) + j;
                int n = n0 + wn * 32 + tj * 16 + (lane & 15);
                float val = fmaxf(scale * acc[ti][tj][j] + bias[n], 0.f);
                val *= (mask[m] != 0) ? 1.f : 0.f;
                out[(size_t)m * HK_ + n] = (__bf16)val;
            }
}

// ---------------- K3: att[b,h,v,q] = sum_k (hmat[h,k]*vp[b,v,k]) * qp[b,q,k] + hbias[h] ------------
// A tile: bf16 vp scaled by h_mat at stage time.  B tile: pure bf16 copy -> async load-to-LDS.
__global__ __launch_bounds__(256) void att_kernel(const __bf16* __restrict__ vp,
                                                  const __bf16* __restrict__ qp,
                                                  const float* __restrict__ h_mat,
                                                  const float* __restrict__ h_bias,
                                                  float* __restrict__ att) {
    __shared__ __bf16 ldsA[2][128 * LDT];
    __shared__ __bf16 ldsB[2][128 * LDT];
    const int b = blockIdx.z, h = blockIdx.y;
    const int v0 = (blockIdx.x & 1) * 128;
    const int q0 = (blockIdx.x >> 1) * 128;
    const int tid = threadIdx.x;
    const int lane = tid & 31, wid = tid >> 5;
    const int wm = wid >> 2, wn = wid & 3;
    const int r = tid >> 1;
    const int half = (tid & 1) * 16;

    const __bf16* vbase = vp + (size_t)b * VN_ * HK_;
    const __bf16* qbase = qp + (size_t)b * QN_ * HK_;
    const float*  hrow  = h_mat + (size_t)h * HK_;

    v8f acc[4][2];
#pragma unroll
    for (int i = 0; i < 4; ++i)
#pragma unroll
        for (int j = 0; j < 2; ++j) acc[i][j] = (v8f){0.f,0.f,0.f,0.f,0.f,0.f,0.f,0.f};

    const int nk = HK_ >> 5;  // 24
    v8bf va0, va1;
    float hs[16];

    auto issueA = [&](int kt) {
        const __bf16* pa = vbase + (size_t)(v0 + r) * HK_ + kt * 32 + half;
        va0 = *(const v8bf*)pa;  va1 = *(const v8bf*)(pa + 8);
        const float* hp = hrow + kt * 32 + half;
#pragma unroll
        for (int i = 0; i < 16; ++i) hs[i] = hp[i];
    };
    auto commitA = [&](int buf) {
        v8bf s0, s1;
#pragma unroll
        for (int i = 0; i < 8; ++i) {
            s0[i] = (__bf16)((float)va0[i] * hs[i]);
            s1[i] = (__bf16)((float)va1[i] * hs[8 + i]);
        }
        *(v8bf*)&ldsA[buf][r * LDT + half]     = s0;
        *(v8bf*)&ldsA[buf][r * LDT + half + 8] = s1;
    };
#if defined(HAVE_ASYNC_LDS)
    auto stageB = [&](int kt, int buf) {  // async global->LDS, no VGPR round trip
        const __bf16* pb = qbase + (size_t)(q0 + r) * HK_ + kt * 32 + half;
        __bf16* dst = &ldsB[buf][r * LDT + half];
        async_copy_b128(pb, dst);
        async_copy_b128(pb + 8, dst + 8);
    };
#else
    auto stageB = [&](int kt, int buf) {
        const __bf16* pb = qbase + (size_t)(q0 + r) * HK_ + kt * 32 + half;
        v8bf b0 = *(const v8bf*)pb, b1 = *(const v8bf*)(pb + 8);
        *(v8bf*)&ldsB[buf][r * LDT + half]     = b0;
        *(v8bf*)&ldsB[buf][r * LDT + half + 8] = b1;
    };
#endif

    issueA(0); commitA(0);
    stageB(0, 0);
#if defined(HAVE_ASYNC_LDS)
    WAIT_ASYNC();
#endif
    __syncthreads();
    for (int kt = 0; kt < nk; ++kt) {
        const int buf = kt & 1;
        const bool more = (kt + 1) < nk;
        if (more) { issueA(kt + 1); stageB(kt + 1, buf ^ 1); }
        v16bf af[4], bfr[2];
#pragma unroll
        for (int ti = 0; ti < 4; ++ti) af[ti] = frag_a(&ldsA[buf][(wm * 64 + ti * 16) * LDT], lane);
#pragma unroll
        for (int tj = 0; tj < 2; ++tj) bfr[tj] = frag_b(&ldsB[buf][(wn * 32 + tj * 16) * LDT], lane);
#pragma unroll
        for (int ti = 0; ti < 4; ++ti)
#pragma unroll
            for (int tj = 0; tj < 2; ++tj)
                acc[ti][tj] = __builtin_amdgcn_wmma_f32_16x16x32_bf16(
                    false, af[ti], false, bfr[tj], (short)0, acc[ti][tj], false, false);
        if (more) commitA(buf ^ 1);
#if defined(HAVE_ASYNC_LDS)
        if (more) WAIT_ASYNC();
#endif
        __syncthreads();
    }

    const float hb = h_bias[h];
    float* obase = att + ((size_t)(b * H_OUT_ + h)) * VN_ * QN_;
#pragma unroll
    for (int ti = 0; ti < 4; ++ti)
#pragma unroll
        for (int tj = 0; tj < 2; ++tj)
#pragma unroll
            for (int j = 0; j < 8; ++j) {
                int v = v0 + wm * 64 + ti * 16 + ((lane >> 4) << 3) + j;
                int qc = q0 + wn * 32 + tj * 16 + (lane & 15);
                obase[(size_t)v * QN_ + qc] = acc[ti][tj][j] + hb;
            }
}

// ---------------- K4: masked softmax over VN*QN per (b,h), in place on d_out ----------------
__global__ __launch_bounds__(256) void softmax_kernel(float* __restrict__ att,
                                                      const int* __restrict__ v_mask,
                                                      const int* __restrict__ q_mask,
                                                      const int* __restrict__ smflag) {
    __shared__ int vm[VN_];
    __shared__ int qm[QN_];
    __shared__ float red[256];
    const int bh = blockIdx.x;
    const int b = bh >> 1;
    const int tid = threadIdx.x;
    float* base = att + (size_t)bh * VN_ * QN_;

    for (int i = tid; i < VN_; i += 256) vm[i] = v_mask[b * VN_ + i];
    for (int i = tid; i < QN_; i += 256) qm[i] = q_mask[b * QN_ + i];
    __syncthreads();

    const int total = VN_ * QN_;
    const int sm = smflag[0];
    if (!sm) {
        for (int i = tid; i < total; i += 256) {
            int v = i >> 10, qc = i & 1023;
            if (!(vm[v] && qm[qc])) base[i] = 0.f;
        }
        return;
    }
    float m = -INFINITY;
    for (int i = tid; i < total; i += 256) {
        int v = i >> 10, qc = i & 1023;
        if (vm[v] && qm[qc]) m = fmaxf(m, base[i]);
    }
    m = block_reduce_max(m, red, tid);
    if (m == -INFINITY) {  // no valid elements: reference yields all-zero att
        for (int i = tid; i < total; i += 256) base[i] = 0.f;
        return;
    }
    float s = 0.f;
    for (int i = tid; i < total; i += 256) {
        int v = i >> 10, qc = i & 1023;
        if (vm[v] && qm[qc]) s += __expf(base[i] - m);
    }
    s = block_reduce_sum(s, red, tid);
    const float inv = 1.f / fmaxf(s, 1e-20f);
    for (int i = tid; i < total; i += 256) {
        int v = i >> 10, qc = i & 1023;
        float x = base[i];
        base[i] = (vm[v] && qm[qc]) ? __expf(x - m) * inv : 0.f;
    }
}

// ---------------- zero helper ----------------
__global__ void zero_kernel(float* __restrict__ p, int n) {
    int i = blockIdx.x * blockDim.x + threadIdx.x;
    if (i < n) p[i] = 0.f;
}

// ---------------- K5: fusion[b,k] += sum_v vp[b,v,k] * (att[b,h] @ qp[b])[v,k] ----------------
// WMMA GEMM: A = att (fp32 -> bf16), B = qp^T (transposed at LDS-stage time), K-dim = q.
__global__ __launch_bounds__(256) void fusion_kernel(const float* __restrict__ att,
                                                     const __bf16* __restrict__ vp,
                                                     const __bf16* __restrict__ qp,
                                                     float* __restrict__ fusion) {
    __shared__ __bf16 ldsA[2][128 * LDT];
    __shared__ __bf16 ldsB[2][128 * LDT];
    __shared__ float colsum[128];
    const int b = blockIdx.z, h = blockIdx.y;
    const int v0  = (blockIdx.x % 2) * 128;   // VN tile
    const int k0c = (blockIdx.x / 2) * 128;   // HK output-column tile
    const int tid = threadIdx.x;
    const int lane = tid & 31, wid = tid >> 5;
    const int wm = wid >> 2, wn = wid & 3;
    const int r = tid >> 1;                   // A loader row
    const int half = (tid & 1) * 16;          // A loader q sub-chunk
    const int qi = tid >> 3;                  // B loader: q row 0..31
    const int kc = (tid & 7) * 16;            // B loader: k chunk 0..112

    const float*  abase = att + ((size_t)(b * H_OUT_ + h)) * VN_ * QN_;
    const __bf16* qbase = qp + (size_t)b * QN_ * HK_;

    v8f acc[4][2];
#pragma unroll
    for (int i = 0; i < 4; ++i)
#pragma unroll
        for (int j = 0; j < 2; ++j) acc[i][j] = (v8f){0.f,0.f,0.f,0.f,0.f,0.f,0.f,0.f};

    const int nk = QN_ >> 5;  // 32
    float fa[16];
    v8bf vb0, vb1;

    auto issue = [&](int qt) {
        const float* pa = abase + (size_t)(v0 + r) * QN_ + qt * 32 + half;
#pragma unroll
        for (int i = 0; i < 16; ++i) fa[i] = pa[i];
        const __bf16* pb = qbase + (size_t)(qt * 32 + qi) * HK_ + k0c + kc;
        vb0 = *(const v8bf*)pb;  vb1 = *(const v8bf*)(pb + 8);
    };
    auto commit = [&](int buf) {
        v8bf a0, a1;
#pragma unroll
        for (int i = 0; i < 8; ++i) { a0[i] = (__bf16)fa[i]; a1[i] = (__bf16)fa[8 + i]; }
        *(v8bf*)&ldsA[buf][r * LDT + half]     = a0;
        *(v8bf*)&ldsA[buf][r * LDT + half + 8] = a1;
#pragma unroll
        for (int i = 0; i < 8; ++i) {          // transpose qp tile: ldsB[k][q]
            ldsB[buf][(kc + i) * LDT + qi]     = vb0[i];
            ldsB[buf][(kc + 8 + i) * LDT + qi] = vb1[i];
        }
    };

    issue(0); commit(0);
    __syncthreads();
    for (int qt = 0; qt < nk; ++qt) {
        const int buf = qt & 1;
        const bool more = (qt + 1) < nk;
        if (more) issue(qt + 1);
        v16bf af[4], bfr[2];
#pragma unroll
        for (int ti = 0; ti < 4; ++ti) af[ti] = frag_a(&ldsA[buf][(wm * 64 + ti * 16) * LDT], lane);
#pragma unroll
        for (int tj = 0; tj < 2; ++tj) bfr[tj] = frag_b(&ldsB[buf][(wn * 32 + tj * 16) * LDT], lane);
#pragma unroll
        for (int ti = 0; ti < 4; ++ti)
#pragma unroll
            for (int tj = 0; tj < 2; ++tj)
                acc[ti][tj] = __builtin_amdgcn_wmma_f32_16x16x32_bf16(
                    false, af[ti], false, bfr[tj], (short)0, acc[ti][tj], false, false);
        if (more) commit(buf ^ 1);
        __syncthreads();
    }

    // Weighted reduction over v (rows): fusion[b, k0c+kloc] += sum_v vp[b,v,k]*t[v,k]
    if (tid < 128) colsum[tid] = 0.f;
    __syncthreads();
    const __bf16* vbase = vp + (size_t)b * VN_ * HK_;
#pragma unroll
    for (int ti = 0; ti < 4; ++ti)
#pragma unroll
        for (int tj = 0; tj < 2; ++tj)
#pragma unroll
            for (int j = 0; j < 8; ++j) {
                int vloc = wm * 64 + ti * 16 + ((lane >> 4) << 3) + j;
                int kloc = wn * 32 + tj * 16 + (lane & 15);
                float w = (float)vbase[(size_t)(v0 + vloc) * HK_ + k0c + kloc];
                atomicAdd(&colsum[kloc], acc[ti][tj][j] * w);  // LDS float atomic (ds path)
            }
    __syncthreads();
    if (tid < 128) atomicAdd(&fusion[(size_t)b * HK_ + k0c + tid], colsum[tid]);
}

// ---------------- K6: group-of-3 sum + LayerNorm ----------------
__global__ __launch_bounds__(256) void head_kernel(const float* __restrict__ fusion,
                                                   const float* __restrict__ gamma,
                                                   const float* __restrict__ beta,
                                                   float* __restrict__ out) {
    __shared__ float red[256];
    const int b = blockIdx.x;
    const int d = threadIdx.x;  // 0..255
    const float* f = fusion + (size_t)b * HK_;
    float x = f[3 * d] + f[3 * d + 1] + f[3 * d + 2];
    float mu = block_reduce_sum(x, red, d) * (1.f / H_DIM_);
    float c = x - mu;
    float var = block_reduce_sum(c * c, red, d) * (1.f / H_DIM_);
    out[(size_t)b * H_DIM_ + d] = c * rsqrtf(var + LN_EPS) * gamma[d] + beta[d];
}

// ---------------- host launcher ----------------
extern "C" void kernel_launch(void* const* d_in, const int* in_sizes, int n_in,
                              void* d_out, int out_size, void* d_ws, size_t ws_size,
                              hipStream_t stream) {
    const float* v      = (const float*)d_in[0];
    const float* q      = (const float*)d_in[1];
    const int*   v_mask = (const int*)d_in[2];
    const int*   q_mask = (const int*)d_in[3];
    const int*   smflag = (const int*)d_in[4];
    const float* v_w    = (const float*)d_in[5];
    const float* v_g    = (const float*)d_in[6];
    const float* v_b    = (const float*)d_in[7];
    const float* q_w    = (const float*)d_in[8];
    const float* q_g    = (const float*)d_in[9];
    const float* q_b    = (const float*)d_in[10];
    const float* h_mat  = (const float*)d_in[11];
    const float* h_bias = (const float*)d_in[12];
    const float* ln_g   = (const float*)d_in[13];
    const float* ln_b   = (const float*)d_in[14];

    float* logits = (float*)d_out;
    float* att    = logits + (size_t)B_ * H_DIM_;

    // workspace layout
    char* ws = (char*)d_ws;
    const size_t vp_bytes = (size_t)B_ * VN_ * HK_ * sizeof(__bf16);   // 25.2 MB
    const size_t qp_bytes = (size_t)B_ * QN_ * HK_ * sizeof(__bf16);   // 100.7 MB
    float*  ssq    = (float*)ws;
    __bf16* vp     = (__bf16*)(ws + 256);
    __bf16* qp     = (__bf16*)(ws + 256 + vp_bytes);
    float*  fusion = (float*)(ws + 256 + vp_bytes + qp_bytes);         // B*HK fp32

    ssq_kernel<<<2, 256, 0, stream>>>(v_w, q_w, ssq);

    proj_kernel<<<dim3(B_ * VN_ / 128, HK_ / 128), 256, 0, stream>>>(
        v, v_w, v_b, v_mask, v_g, ssq + 0, V_DIM_, vp);
    proj_kernel<<<dim3(B_ * QN_ / 128, HK_ / 128), 256, 0, stream>>>(
        q, q_w, q_b, q_mask, q_g, ssq + 1, Q_DIM_, qp);

    att_kernel<<<dim3((VN_ / 128) * (QN_ / 128), H_OUT_, B_), 256, 0, stream>>>(
        vp, qp, h_mat, h_bias, att);

    softmax_kernel<<<B_ * H_OUT_, 256, 0, stream>>>(att, v_mask, q_mask, smflag);

    zero_kernel<<<(B_ * HK_ + 255) / 256, 256, 0, stream>>>(fusion, B_ * HK_);

    fusion_kernel<<<dim3((VN_ / 128) * (HK_ / 128), H_OUT_, B_), 256, 0, stream>>>(
        att, vp, qp, fusion);

    head_kernel<<<B_, H_DIM_, 0, stream>>>(fusion, ln_g, ln_b, logits);
}
